// TADGAT_50895362458274
// MI455X (gfx1250) — matrix operand order repaired and checked by reading, not compiled, not run on hardware
//
#include <hip/hip_runtime.h>
#include <hip/hip_bf16.h>
#include <math.h>

// ---------------------------------------------------------------------------
// TADGAT on MI455X (gfx1250, wave32, WMMA)
// GEMMs use v_wmma_f32_16x16x32_bf16 with BOTH operands read as two 16-byte
// chunks per lane (k 8h..+7, 16+8h..+7) from global (weights are pre-transposed to column-major
// during the f32->bf16 conversion pass). Edge phase uses L2-resident table
// gathers + float atomics. Bandwidth-bound overall (~0.6 GB @ 23.3 TB/s).
// ---------------------------------------------------------------------------

typedef __attribute__((ext_vector_type(16))) __bf16 v16bf;
typedef __attribute__((ext_vector_type(8)))  __bf16 v8bf;
typedef __attribute__((ext_vector_type(8)))  float  v8f;

#define NEG_SLOPE 0.2f
#define ENC_NEG_INF 0x007FFFFFu   // ordered-uint encoding of -inf

__device__ __forceinline__ unsigned short f32_to_bf16_rne(float f) {
  unsigned int u = __float_as_uint(f);
  u += 0x7fffu + ((u >> 16) & 1u);
  return (unsigned short)(u >> 16);
}
__device__ __forceinline__ float leakyf(float x) { return x > 0.f ? x : NEG_SLOPE * x; }

// monotonic float<->uint encoding so unsigned atomicMax == float max
__device__ __forceinline__ unsigned int enc_f32(float f) {
  unsigned int u = __float_as_uint(f);
  return (u & 0x80000000u) ? ~u : (u | 0x80000000u);
}
__device__ __forceinline__ float dec_f32(unsigned int e) {
  unsigned int u = (e & 0x80000000u) ? (e & 0x7fffffffu) : ~e;
  return __uint_as_float(u);
}

// ------------------------------- utility kernels ---------------------------

__global__ void fill_u32_kernel(unsigned int* p, unsigned int v, int n) {
  int i = blockIdx.x * blockDim.x + threadIdx.x;
  if (i < n) p[i] = v;
}
__global__ void fill_f32_kernel(float* p, float v, int n) {
  int i = blockIdx.x * blockDim.x + threadIdx.x;
  if (i < n) p[i] = v;
}
__global__ void f32_to_bf16_kernel(unsigned short* o, const float* __restrict__ in, int n) {
  int i = blockIdx.x * blockDim.x + threadIdx.x;
  if (i < n) o[i] = f32_to_bf16_rne(in[i]);
}

// Transpose + convert: in[K,N] f32 row-major -> o[N,K] bf16 row-major.
// LDS 32x32 tile (padded), coalesced on both sides.
__global__ __launch_bounds__(256)
void f32_to_bf16_tr_kernel(unsigned short* __restrict__ o, const float* __restrict__ in,
                           int K, int N) {
  __shared__ float tile[32][33];
  const int k0 = blockIdx.y << 5, n0 = blockIdx.x << 5;
  const int tx = threadIdx.x, ty = threadIdx.y;   // 32 x 8
#pragma unroll
  for (int s = 0; s < 32; s += 8) {
    int k = k0 + ty + s, n = n0 + tx;
    if (k < K && n < N) tile[ty + s][tx] = in[(size_t)k * N + n];
  }
  __syncthreads();
#pragma unroll
  for (int s = 0; s < 32; s += 8) {
    int n = n0 + ty + s, k = k0 + tx;
    if (n < N && k < K) o[(size_t)n * K + k] = f32_to_bf16_rne(tile[tx][ty + s]);
  }
}

__global__ void relu_kernel(float* o, const float* __restrict__ in, int n) {
  int i = blockIdx.x * blockDim.x + threadIdx.x;
  if (i < n) o[i] = fmaxf(in[i], 0.f);
}

// x[n, 0:128] = id_emb[node_idx[n], 0:128]
__global__ void gather_x_kernel(float* x, const float* __restrict__ emb,
                                const int* __restrict__ idx, int n128) {
  int i = blockIdx.x * blockDim.x + threadIdx.x;
  if (i >= n128) return;
  int n = i >> 7, c = i & 127;
  x[i] = emb[(size_t)idx[n] * 128 + c];
}

// feat[b] = [x0[cn] | x1[cn] | x2[cn]]   (B x 384)
__global__ void feat_gather_kernel(float* feat, const float* __restrict__ x0,
                                   const float* __restrict__ x1,
                                   const float* __restrict__ x2,
                                   const int* __restrict__ cn, int total) {
  int i = blockIdx.x * blockDim.x + threadIdx.x;
  if (i >= total) return;
  int b = i / 384, j = i - b * 384;
  size_t n = (size_t)cn[b] * 128;
  float v = (j < 128) ? x0[n + j] : (j < 256) ? x1[n + j - 128] : x2[n + j - 256];
  feat[i] = v;
}

// out[n,h] = dot(feat[n, h*32 : h*32+32], avec[h, :])
__global__ void node_dot_kernel(float* out, const float* __restrict__ feat,
                                const float* __restrict__ avec, int n4) {
  int i = blockIdx.x * blockDim.x + threadIdx.x;
  if (i >= n4) return;
  int n = i >> 2, h = i & 3;
  const float* f = feat + (size_t)n * 128 + h * 32;
  const float* a = avec + h * 32;
  float s = 0.f;
#pragma unroll
  for (int c = 0; c < 32; ++c) s += f[c] * a[c];
  out[i] = s;
}

// ------------------------------- edge kernels ------------------------------

// logit[e,h] = leaky( ssrc[src] + dot(att_bias+att_dist, asrc[h]) + sdst[dst] )
// + segment-max into menc[dst,h] via ordered-uint atomicMax
__global__ void edge_logit_kernel(float* __restrict__ logit, unsigned int* __restrict__ menc,
                                  const float* __restrict__ ssrc, const float* __restrict__ sdst,
                                  const int* __restrict__ src, const int* __restrict__ dst,
                                  const int* __restrict__ etype, const int* __restrict__ toff,
                                  const int* __restrict__ dists,
                                  const float* __restrict__ u_att, const float* __restrict__ i_att,
                                  const float* __restrict__ tr_att,
                                  const float* __restrict__ ud_att, const float* __restrict__ id_att,
                                  const float* __restrict__ asrc, int e4) {
  int i = blockIdx.x * blockDim.x + threadIdx.x;
  if (i >= e4) return;
  int e = i >> 2, h = i & 3;
  int t = etype[e];
  bool isu = (t == 0), ishomo = (t < 2);
  float dscale = ((t == 0) | (t == 2) | (t == 3)) ? 1.f : 0.f;
  const float* btab = isu ? u_att : (ishomo ? i_att : tr_att);
  const float* dtab = isu ? ud_att : id_att;
  const float* b = btab + (size_t)toff[e] * 128 + h * 32;
  const float* d = dtab + (size_t)dists[e] * 128 + h * 32;
  const float* a = asrc + h * 32;
  float s = 0.f;
#pragma unroll
  for (int c = 0; c < 32; ++c) s += (b[c] + dscale * d[c]) * a[c];
  int de = dst[e];
  float lg = leakyf(ssrc[src[e] * 4 + h] + s + sdst[de * 4 + h]);
  logit[i] = lg;
  atomicMax(&menc[de * 4 + h], enc_f32(lg));
}

// a[e,h] = exp(logit - m[dst,h]); den[dst,h] += a
__global__ void edge_exp_kernel(float* __restrict__ aexp, float* __restrict__ den,
                                const float* __restrict__ logit,
                                const unsigned int* __restrict__ menc,
                                const int* __restrict__ dst, int e4) {
  int i = blockIdx.x * blockDim.x + threadIdx.x;
  if (i >= e4) return;
  int e = i >> 2, h = i & 3;
  int de = dst[e];
  float v = __expf(logit[i] - dec_f32(menc[de * 4 + h]));
  aexp[i] = v;
  atomicAdd(&den[de * 4 + h], v);
}

// out[dst,c] += alpha * (k[src,c] + emb_bias + emb_dist)
__global__ void edge_accum_kernel(float* __restrict__ out, const float* __restrict__ aexp,
                                  const float* __restrict__ den, const float* __restrict__ k,
                                  const int* __restrict__ src, const int* __restrict__ dst,
                                  const int* __restrict__ etype, const int* __restrict__ toff,
                                  const int* __restrict__ dists,
                                  const float* __restrict__ u_emb, const float* __restrict__ i_emb,
                                  const float* __restrict__ tr_emb,
                                  const float* __restrict__ ud_emb, const float* __restrict__ id_emb_t,
                                  int e128) {
  int i = blockIdx.x * blockDim.x + threadIdx.x;
  if (i >= e128) return;
  int e = i >> 7, c = i & 127, h = c >> 5;
  int t = etype[e];
  bool isu = (t == 0), ishomo = (t < 2);
  float dscale = ((t == 0) | (t == 2) | (t == 3)) ? 1.f : 0.f;
  const float* btab = isu ? u_emb : (ishomo ? i_emb : tr_emb);
  const float* dtab = isu ? ud_emb : id_emb_t;
  float b = btab[(size_t)toff[e] * 128 + c];
  float d = dscale * dtab[(size_t)dists[e] * 128 + c];
  int de = dst[e];
  float alpha = aexp[e * 4 + h] / fmaxf(den[de * 4 + h], 1e-16f);
  float msg = k[(size_t)src[e] * 128 + c] + b + d;
  atomicAdd(&out[(size_t)de * 128 + c], alpha * msg);
}

// ------------------------------- WMMA GEMM ---------------------------------
// C[M,N](f32) (+= existing C if accum) = A[M,K](bf16 row-major) x BT (bf16,
// COLUMN-major of B[K,N], i.e. BT[N,K] row-major), +bias[n], then activation.
// Requires M%16==0, N%16==0, K%32==0 (true for every call here).
// One wave per 16x16 C tile; both fragments are two 16-byte loads per lane, so
// the inner loop is pure global_load_b128 + v_wmma (no LDS, no barriers).
__global__ __launch_bounds__(256)
void gemm_bf16_wmma_kernel(const unsigned short* __restrict__ Abuf,
                           const unsigned short* __restrict__ BTbuf,
                           float* __restrict__ C, const float* __restrict__ bias,
                           int M, int N, int K, int act, int accum) {
  const __bf16* A  = (const __bf16*)Abuf;
  const __bf16* BT = (const __bf16*)BTbuf;

  const int lane = threadIdx.x;               // 0..31
  const int wave = threadIdx.y;               // 0..7
  const int mt = blockIdx.y * 8 + wave;
  if (mt >= (M >> 4)) return;                 // no barriers -> safe early-out
  const int m0 = mt << 4;
  const int n0 = blockIdx.x << 4;
  const int half = lane >> 4;                 // 0: lanes 0-15, 1: lanes 16-31
  const int r = lane & 15;

  // A fragment rows: lane r -> row m0+r; K elems kk+half*8..+7 and kk+16+half*8..+7
  const __bf16* ap = A + (size_t)(m0 + r) * K + half * 8;
  // B fragment cols: lane r -> col n0+r; K elems kk+half*8..+7 and kk+16+half*8..+7
  const __bf16* bp = BT + (size_t)(n0 + r) * K + half * 8;

  v8f acc = {};
  for (int kk = 0; kk < K; kk += 32) {
    __builtin_prefetch((const void*)(ap + kk + 32), 0, 1);
    __builtin_prefetch((const void*)(bp + kk + 32), 0, 1);
    v8bf alo = *(const v8bf*)(ap + kk);
    v8bf ahi = *(const v8bf*)(ap + kk + 16);
    v16bf af;
#pragma unroll
    for (int j = 0; j < 8; ++j) { af[j] = alo[j]; af[8 + j] = ahi[j]; }
    v8bf blo = *(const v8bf*)(bp + kk);
    v8bf bhi = *(const v8bf*)(bp + kk + 16);
    v16bf bf;
#pragma unroll
    for (int j = 0; j < 8; ++j) { bf[j] = blo[j]; bf[8 + j] = bhi[j]; }
    acc = __builtin_amdgcn_wmma_f32_16x16x32_bf16(false, af, false, bf,
                                                  (short)0, acc, false, false);
  }

  const int nC = n0 + r;
  const float bv = bias ? bias[nC] : 0.f;
#pragma unroll
  for (int i = 0; i < 8; ++i) {
    size_t idx = (size_t)(m0 + i + half * 8) * N + nC;
    float v = acc[i] + bv;
    if (accum) v += C[idx];
    if (act == 1)      v = fmaxf(v, 0.f);
    else if (act == 2) v = leakyf(v);
    else if (act == 3) v = tanhf(v);
    C[idx] = v;
  }
}

// ------------------------------- host driver -------------------------------

extern "C" void kernel_launch(void* const* d_in, const int* in_sizes, int n_in,
                              void* d_out, int out_size, void* d_ws, size_t ws_size,
                              hipStream_t stream) {
  const int N = in_sizes[0];        // 50000 nodes
  const int E = in_sizes[2];        // 200000 edges
  const int B = in_sizes[7];        // 1024
  const int ITEM = in_sizes[34];    // 50000

  const int*   node_idx = (const int*)d_in[0];
  const int*   e_src    = (const int*)d_in[1];
  const int*   e_dst    = e_src + E;
  const int*   e_type   = (const int*)d_in[2];
  const int*   toff0    = (const int*)d_in[3];
  const int*   toff1    = (const int*)d_in[4];
  const int*   dists0   = (const int*)d_in[5];
  const int*   dists1   = (const int*)d_in[6];
  const int*   center   = (const int*)d_in[7];
  const float* id_emb   = (const float*)d_in[8];
  const float* uc_att   = (const float*)d_in[9];
  const float* ic_att   = (const float*)d_in[10];
  const float* tr_att   = (const float*)d_in[11];
  const float* uc_emb   = (const float*)d_in[12];
  const float* ic_emb   = (const float*)d_in[13];
  const float* tr_emb   = (const float*)d_in[14];
  const float* uc_att2  = (const float*)d_in[15];
  const float* ic_att2  = (const float*)d_in[16];
  const float* tr_att2  = (const float*)d_in[17];
  const float* uc_emb2  = (const float*)d_in[18];
  const float* ic_emb2  = (const float*)d_in[19];
  const float* tr_emb2  = (const float*)d_in[20];
  const float* ud_att   = (const float*)d_in[21];
  const float* id_att   = (const float*)d_in[22];
  const float* ud_emb   = (const float*)d_in[23];
  const float* id_embt  = (const float*)d_in[24];
  const float* ud_att2  = (const float*)d_in[25];
  const float* id_att2  = (const float*)d_in[26];
  const float* ud_emb2  = (const float*)d_in[27];
  const float* id_emb2  = (const float*)d_in[28];
  const float* trans_head = (const float*)d_in[29];
  const float* rnn_w    = (const float*)d_in[30];
  const float* w1       = (const float*)d_in[31];
  const float* b1       = (const float*)d_in[32];
  const float* w2       = (const float*)d_in[33];
  const float* b2       = (const float*)d_in[34];
  const float* wl0      = (const float*)d_in[35];
  const float* wr0      = (const float*)d_in[36];
  const float* asrc0    = (const float*)d_in[37];
  const float* adst0    = (const float*)d_in[38];
  const float* wl1      = (const float*)d_in[39];
  const float* wr1      = (const float*)d_in[40];
  const float* asrc1    = (const float*)d_in[41];
  const float* adst1    = (const float*)d_in[42];

  // ---- workspace carve-out (256-B aligned) ----
  const size_t NF = (size_t)N * 128;
  size_t off = 0;
  auto carve = [&](size_t bytes) { size_t o = off; off += (bytes + 255) & ~(size_t)255; return o; };
  char* w = (char*)d_ws;
  float* x0   = (float*)(w + carve(NF * 4));
  float* x1   = (float*)(w + carve(NF * 4));
  float* x2   = (float*)(w + carve(NF * 4));
  float* wA   = (float*)(w + carve(NF * 4));   // layer0: last; layer1: relu(x1)
  float* kbuf = (float*)(w + carve(NF * 4));
  float* qbuf = (float*)(w + carve(NF * 4));
  float* conv = (float*)(w + carve(NF * 4));
  unsigned short* abf = (unsigned short*)(w + carve(NF * 2));          // A operand bf16
  unsigned short* wbf = (unsigned short*)(w + carve((size_t)128 * ITEM * 2)); // BT bf16
  float*        logitE = (float*)(w + carve((size_t)E * 4 * 4));
  float*        aE     = (float*)(w + carve((size_t)E * 4 * 4));
  unsigned int* menc   = (unsigned int*)(w + carve((size_t)N * 4 * 4));
  float*        den    = (float*)(w + carve((size_t)N * 4 * 4));
  float*        ssrc   = (float*)(w + carve((size_t)N * 4 * 4));
  float*        sdst   = (float*)(w + carve((size_t)N * 4 * 4));
  float*        feat   = (float*)(w + carve((size_t)B * 384 * 4));
  float*        hdd    = (float*)(w + carve((size_t)B * 128 * 4));
  (void)ws_size;

  auto blocks = [](long n) { return dim3((unsigned)((n + 255) / 256)); };
  auto cvt = [&](unsigned short* o, const float* in, long n) {
    f32_to_bf16_kernel<<<blocks(n), 256, 0, stream>>>(o, in, (int)n);
  };
  // transpose-convert weight W[K,N] -> WT[N,K] bf16
  auto cvt_tr = [&](unsigned short* o, const float* in, int K, int Nc) {
    dim3 blk(32, 8, 1);
    dim3 grd((unsigned)((Nc + 31) / 32), (unsigned)((K + 31) / 32), 1);
    f32_to_bf16_tr_kernel<<<grd, blk, 0, stream>>>(o, in, K, Nc);
  };
  auto gemm = [&](const unsigned short* A, const unsigned short* BT, float* C,
                  const float* bias, int M, int Nc, int K, int act, int accum) {
    dim3 blk(32, 8, 1);
    dim3 grd((unsigned)(Nc / 16), (unsigned)((M / 16 + 7) / 8), 1);
    gemm_bf16_wmma_kernel<<<grd, blk, 0, stream>>>(A, BT, C, bias, M, Nc, K, act, accum);
  };

  // ---- gather node embeddings ----
  gather_x_kernel<<<blocks(NF), 256, 0, stream>>>(x0, id_emb, node_idx, (int)NF);

  // ======================= layer 0 (input h = x0) =======================
  cvt(abf, x0, NF);
  cvt_tr(wbf, trans_head, 128, 128); gemm(abf, wbf, wA,   nullptr, N, 128, 128, 0, 0); // last
  cvt_tr(wbf, wl0,        128, 128); gemm(abf, wbf, kbuf, nullptr, N, 128, 128, 0, 0);
  cvt_tr(wbf, wr0,        128, 128); gemm(abf, wbf, qbuf, nullptr, N, 128, 128, 0, 0);

  node_dot_kernel<<<blocks((long)N * 4), 256, 0, stream>>>(ssrc, kbuf, asrc0, N * 4);
  node_dot_kernel<<<blocks((long)N * 4), 256, 0, stream>>>(sdst, qbuf, adst0, N * 4);
  fill_u32_kernel<<<blocks((long)N * 4), 256, 0, stream>>>(menc, ENC_NEG_INF, N * 4);
  fill_f32_kernel<<<blocks((long)N * 4), 256, 0, stream>>>(den, 0.f, N * 4);
  fill_f32_kernel<<<blocks(NF), 256, 0, stream>>>(conv, 0.f, (int)NF);

  edge_logit_kernel<<<blocks((long)E * 4), 256, 0, stream>>>(
      logitE, menc, ssrc, sdst, e_src, e_dst, e_type, toff0, dists0,
      uc_att, ic_att, tr_att, ud_att, id_att, asrc0, E * 4);
  edge_exp_kernel<<<blocks((long)E * 4), 256, 0, stream>>>(aE, den, logitE, menc, e_dst, E * 4);
  edge_accum_kernel<<<blocks((long)E * 128), 256, 0, stream>>>(
      conv, aE, den, kbuf, e_src, e_dst, e_type, toff0, dists0,
      uc_emb, ic_emb, tr_emb, ud_emb, id_embt, E * 128);

  // x1 = tanh(conv @ rnn_w[0:128] + last @ rnn_w[128:256])   (split-K, accum)
  cvt_tr(wbf,               rnn_w,             128, 128);
  cvt_tr(wbf + 128 * 128,   rnn_w + 128 * 128, 128, 128);
  cvt(abf, conv, NF); gemm(abf, wbf,             x1, nullptr, N, 128, 128, 0, 0);
  cvt(abf, wA,   NF); gemm(abf, wbf + 128 * 128, x1, nullptr, N, 128, 128, 3, 1);

  // ======================= layer 1 (input h = relu(x1)) =======================
  relu_kernel<<<blocks(NF), 256, 0, stream>>>(wA, x1, (int)NF);
  cvt(abf, wA, NF);
  cvt_tr(wbf, wl1, 128, 128); gemm(abf, wbf, kbuf, nullptr, N, 128, 128, 0, 0);
  cvt_tr(wbf, wr1, 128, 128); gemm(abf, wbf, qbuf, nullptr, N, 128, 128, 0, 0);

  node_dot_kernel<<<blocks((long)N * 4), 256, 0, stream>>>(ssrc, kbuf, asrc1, N * 4);
  node_dot_kernel<<<blocks((long)N * 4), 256, 0, stream>>>(sdst, qbuf, adst1, N * 4);
  fill_u32_kernel<<<blocks((long)N * 4), 256, 0, stream>>>(menc, ENC_NEG_INF, N * 4);
  fill_f32_kernel<<<blocks((long)N * 4), 256, 0, stream>>>(den, 0.f, N * 4);
  fill_f32_kernel<<<blocks(NF), 256, 0, stream>>>(conv, 0.f, (int)NF);

  edge_logit_kernel<<<blocks((long)E * 4), 256, 0, stream>>>(
      logitE, menc, ssrc, sdst, e_src, e_dst, e_type, toff1, dists1,
      uc_att2, ic_att2, tr_att2, ud_att2, id_att2, asrc1, E * 4);
  edge_exp_kernel<<<blocks((long)E * 4), 256, 0, stream>>>(aE, den, logitE, menc, e_dst, E * 4);
  edge_accum_kernel<<<blocks((long)E * 128), 256, 0, stream>>>(
      conv, aE, den, kbuf, e_src, e_dst, e_type, toff1, dists1,
      uc_emb2, ic_emb2, tr_emb2, ud_emb2, id_emb2, E * 128);

  cvt_tr(wbf,             rnn_w,             128, 128);
  cvt_tr(wbf + 128 * 128, rnn_w + 128 * 128, 128, 128);
  cvt(abf, conv, NF); gemm(abf, wbf,             x2, nullptr, N, 128, 128, 0, 0);
  cvt(abf, wA,   NF); gemm(abf, wbf + 128 * 128, x2, nullptr, N, 128, 128, 3, 1);

  // ======================= prediction head =======================
  feat_gather_kernel<<<blocks((long)B * 384), 256, 0, stream>>>(feat, x0, x1, x2, center, B * 384);
  cvt(abf, feat, (long)B * 384);
  cvt_tr(wbf, w1, 384, 128);
  gemm(abf, wbf, hdd, b1, B, 128, 384, 2, 0);                 // leaky_relu
  cvt(abf, hdd, (long)B * 128);
  cvt_tr(wbf, w2, 128, ITEM);
  gemm(abf, wbf, (float*)d_out, b2, B, ITEM, 128, 0, 0);      // logits
}
